// BottleneckTPP_26027501814495
// MI455X (gfx1250) — compile-verified, matrix operand order checked by reading, not run on hardware
//
#include <hip/hip_runtime.h>
#include <hip/hip_bf16.h>

// ---------------------------------------------------------------------------
// Types
// ---------------------------------------------------------------------------
typedef __bf16 bf16_t;
typedef __attribute__((ext_vector_type(16))) __bf16 bfrag16;   // WMMA A/B fragment
typedef __attribute__((ext_vector_type(8)))  float  facc8;     // WMMA C/D fragment
typedef unsigned int u32x4 __attribute__((ext_vector_type(4)));
typedef float  f32x4 __attribute__((ext_vector_type(4)));
typedef __bf16 bf4   __attribute__((ext_vector_type(4)));

#define LDS_PITCH 40   // bf16 elems per row = 80 bytes -> conflict-free b128 frag loads

// ---------------------------------------------------------------------------
// Small helpers
// ---------------------------------------------------------------------------
__device__ __forceinline__ float load1f(const float* p)  { return *p; }
__device__ __forceinline__ float load1f(const bf16_t* p) { return (float)*p; }

// Explicit pair conversion: one v_cvt_pk_bf16_f32 per 2 floats (a->lo, b->hi);
// avoids clang's scalar-cvt + v_mov_b16 merge sequence.
__device__ __forceinline__ unsigned cvt_pk_bf16(float a, float b) {
  unsigned r;
  asm("v_cvt_pk_bf16_f32 %0, %1, %2" : "=v"(r) : "v"(a), "v"(b));
  return r;
}
__device__ __forceinline__ u32x4 pack_bf16x8(const float* r) {
  u32x4 q;
  q[0] = cvt_pk_bf16(r[0], r[1]);
  q[1] = cvt_pk_bf16(r[2], r[3]);
  q[2] = cvt_pk_bf16(r[4], r[5]);
  q[3] = cvt_pk_bf16(r[6], r[7]);
  return q;
}

// Fragment load: 2 x ds_load_b128 per 16x32 bf16 fragment; 80B row pitch keeps
// all 64 LDS banks busy with zero conflicts across the 16-lane groups.
__device__ __forceinline__ bfrag16 load_frag(const bf16_t* rowBase, int half) {
  union { u32x4 q[2]; bfrag16 v; } f;
  const u32x4* p = (const u32x4*)(rowBase + half * 8);
  f.q[0] = p[0];        // +0  bytes
  f.q[1] = p[2];        // +32 bytes
  return f.v;
}

// ---------------------------------------------------------------------------
// Stage 1: issue global loads into registers (no waits consumed here).
// Each thread owns one LDS row (m for A, n for B) and a 16-wide K segment:
//   A[m][K]  : 4 contiguous b128 loads along K
//   B[K][n]  : 16 K-strided b32/u16 loads (coalesced across lanes: one K-row,
//              32 consecutive columns per instruction) -> transpose is free.
// N edge: clamp the column, mask at cvt time (keeps waits late).
// ---------------------------------------------------------------------------
template <typename TB>
__device__ __forceinline__ void stage_global(const float* __restrict__ A,
                                             const TB* __restrict__ B,
                                             int K, int Nn, int mBase, int nBase, int k0,
                                             int tid, float (&ra)[16], float (&rb)[16])
{
  const int row = tid & 127;
  const int kh  = (tid >> 7) << 4;          // 0 or 16
  const float* pa = A + (size_t)(mBase + row) * K + k0 + kh;
#pragma unroll
  for (int j = 0; j < 4; ++j) {
    f32x4 v = *(const f32x4*)(pa + j * 4);
    ra[j * 4 + 0] = v[0]; ra[j * 4 + 1] = v[1];
    ra[j * 4 + 2] = v[2]; ra[j * 4 + 3] = v[3];
  }
  const int gc = nBase + row;
  const TB* pb = B + (size_t)(k0 + kh) * Nn + ((gc < Nn) ? gc : 0);
#pragma unroll
  for (int j = 0; j < 16; ++j) rb[j] = load1f(pb + (size_t)j * Nn);
}

// Stage 2: wait (implicit), convert to bf16, 2 x ds_store_b128 per tile row.
__device__ __forceinline__ void stage_lds(bf16_t* lA, bf16_t* lB, int tid,
                                          int Nn, int nBase,
                                          const float (&ra)[16], const float (&rb)[16])
{
  const int row = tid & 127;
  const int kh  = (tid >> 7) << 4;

  u32x4* dA = (u32x4*)(lA + row * LDS_PITCH + kh);
  dA[0] = pack_bf16x8(&ra[0]);
  dA[1] = pack_bf16x8(&ra[8]);

  const bool okB = (nBase + row) < Nn;
  u32x4 q0 = pack_bf16x8(&rb[0]);
  u32x4 q1 = pack_bf16x8(&rb[8]);
  if (!okB) { q0 = 0; q1 = 0; }
  u32x4* dB = (u32x4*)(lB + row * LDS_PITCH + kh);
  dB[0] = q0;
  dB[1] = q1;
}

// ---------------------------------------------------------------------------
// WMMA GEMM + fused per-channel sum / sum-of-squares (BN batch statistics).
// Grid: (ceil(Nn/128), (Mtot/128)*KSPLIT, images). Block: 256 = 8 waves.
// Wave w owns a 32(M) x 64(N) sub-tile -> 2x4 WMMA 16x16 accumulators.
// Software pipeline: global loads for chunk c+1 issue before the WMMAs of
// chunk c; loadcnt waits + cvt + ds_store overlap the matrix pipe.
// KSPLIT>1: blockIdx.y also carries a K-slice; partial results are combined
// with f32 atomics into (pre-zeroed) Y and stats are computed separately.
// ---------------------------------------------------------------------------
template <typename TB, int KSPLIT>
__global__ __launch_bounds__(256)
void wmma_gemm_stats(const float* __restrict__ A, const TB* __restrict__ B,
                     float* __restrict__ Y,
                     float* __restrict__ sSum, float* __restrict__ sSq,
                     int K, int Nn)
{
  __shared__ __align__(16) bf16_t lA[2][128 * LDS_PITCH];
  __shared__ __align__(16) bf16_t lB[2][128 * LDS_PITCH];

  const int tid   = threadIdx.x;
  const int lane  = tid & 31;
  const int wave  = tid >> 5;
  const int l16   = lane & 15;
  const int half  = lane >> 4;
  const int waveM = (wave & 3) * 32;      // 4 waves down M
  const int waveN = (wave >> 2) * 64;     // 2 waves across N

  const int img   = blockIdx.z;
  const int myb   = blockIdx.y / KSPLIT;  // M block
  const int ks    = blockIdx.y - myb * KSPLIT;
  const int mBase = myb * 128;
  const int nBase = blockIdx.x * 128;
  const int Mtot  = (gridDim.y / KSPLIT) * 128;

  const TB* Bimg = B + (size_t)img * K * Nn;

  facc8 acc[2][4] = {};
  float ra[16], rb[16];

  const int chunksPer = (K >> 5) / KSPLIT;
  const int cBeg = ks * chunksPer;
  const int cEnd = cBeg + chunksPer;

  stage_global(A, Bimg, K, Nn, mBase, nBase, cBeg << 5, tid, ra, rb);
  stage_lds(lA[0], lB[0], tid, Nn, nBase, ra, rb);
  __syncthreads();

  for (int c = cBeg; c < cEnd; ++c) {
    const int  cur  = (c - cBeg) & 1;
    const bool more = (c + 1 < cEnd);

    if (more)   // issue next chunk's global loads; don't wait yet
      stage_global(A, Bimg, K, Nn, mBase, nBase, (c + 1) << 5, tid, ra, rb);

    bfrag16 fa[2], fb[4];
#pragma unroll
    for (int mi = 0; mi < 2; ++mi)
      fa[mi] = load_frag(&lA[cur][(waveM + mi * 16 + l16) * LDS_PITCH], half);
#pragma unroll
    for (int ni = 0; ni < 4; ++ni)
      fb[ni] = load_frag(&lB[cur][(waveN + ni * 16 + l16) * LDS_PITCH], half);

#pragma unroll
    for (int mi = 0; mi < 2; ++mi)
#pragma unroll
      for (int ni = 0; ni < 4; ++ni)
        acc[mi][ni] = __builtin_amdgcn_wmma_f32_16x16x32_bf16(
            false, fa[mi], false, fb[ni], (short)0, acc[mi][ni], false, false);

    if (more)   // wait for globals, convert, store into the other LDS buffer
      stage_lds(lA[cur ^ 1], lB[cur ^ 1], tid, Nn, nBase, ra, rb);

    __syncthreads();
  }

  // Epilogue.
  float* Yimg = Y + (size_t)img * Mtot * Nn + (size_t)mBase * Nn;
#pragma unroll
  for (int mi = 0; mi < 2; ++mi) {
#pragma unroll
    for (int r = 0; r < 8; ++r) {
      const int chLoc = waveM + mi * 16 + half * 8 + r;
      if (KSPLIT == 1) {
        float s = 0.0f, q = 0.0f;
#pragma unroll
        for (int ni = 0; ni < 4; ++ni) {
          float v   = acc[mi][ni][r];
          int   col = nBase + waveN + ni * 16 + l16;
          if (col < Nn) Yimg[(size_t)chLoc * Nn + col] = v;
          s += v;
          q += v * v;
        }
#pragma unroll
        for (int off = 8; off > 0; off >>= 1) {           // reduce 16-lane group
          s += __shfl_xor(s, off, 32);
          q += __shfl_xor(q, off, 32);
        }
        if (l16 == 0) {
          atomicAdd(&sSum[mBase + chLoc], s);
          atomicAdd(&sSq [mBase + chLoc], q);
        }
      } else {
#pragma unroll
        for (int ni = 0; ni < 4; ++ni) {
          float v   = acc[mi][ni][r];
          int   col = nBase + waveN + ni * 16 + l16;
          if (col < Nn) atomicAdd(&Yimg[(size_t)chLoc * Nn + col], v);
        }
      }
    }
  }
}

// ---------------------------------------------------------------------------
// Support kernels
// ---------------------------------------------------------------------------
__global__ void zero_kernel(float* p, int n) {
  int i = blockIdx.x * blockDim.x + threadIdx.x;
  if (i < n) p[i] = 0.0f;
}

__global__ __launch_bounds__(256)
void zero4_kernel(float* p, unsigned total4) {
  unsigned t = blockIdx.x * blockDim.x + threadIdx.x;
  if (t < total4) *(f32x4*)(p + (size_t)t * 4) = 0;
}

// Deterministic per-channel sum/sumsq over y[img][C][HW] (for K-split layers).
__global__ __launch_bounds__(256)
void stats_from_y(const float* __restrict__ y, float* __restrict__ sSum,
                  float* __restrict__ sSq, int C, int HW, int IMG)
{
  const int c    = blockIdx.x;
  const int tid  = threadIdx.x;
  const int lane = tid & 31;
  const int wave = tid >> 5;
  float s = 0.0f, q = 0.0f;
  for (int img = 0; img < IMG; ++img) {
    const float* p = y + ((size_t)img * C + c) * HW;
    for (int i = tid; i < HW; i += 256) { float v = p[i]; s += v; q += v * v; }
  }
#pragma unroll
  for (int off = 16; off > 0; off >>= 1) {
    s += __shfl_xor(s, off, 32);
    q += __shfl_xor(q, off, 32);
  }
  __shared__ float ls[8], lq[8];
  if (lane == 0) { ls[wave] = s; lq[wave] = q; }
  __syncthreads();
  if (tid == 0) {
    float S = 0.0f, Q = 0.0f;
#pragma unroll
    for (int w = 0; w < 8; ++w) { S += ls[w]; Q += lq[w]; }
    sSum[c] = S;
    sSq[c]  = Q;
  }
}

__global__ void finalize_bn(const float* __restrict__ sSum, const float* __restrict__ sSq,
                            const float* __restrict__ gamma, const float* __restrict__ beta,
                            float* __restrict__ scale, float* __restrict__ shift,
                            int C, float invCnt)
{
  int c = blockIdx.x * blockDim.x + threadIdx.x;
  if (c >= C) return;
  float mean = sSum[c] * invCnt;
  float var  = sSq[c] * invCnt - mean * mean;
  float inv  = rsqrtf(var + 1e-5f);
  float s    = gamma[c] * inv;
  scale[c] = s;
  shift[c] = beta[c] - mean * s;
}

// BN + ReLU + f32->bf16, 4 elements/thread (b128 in, b64 out). HW % 4 == 0 so
// all 4 elements share one channel; HW/C are compile-time -> mul-shift divides.
template <int HW, int C>
__global__ __launch_bounds__(256)
void bn_relu_store_bf16(const float* __restrict__ y,
                        const float* __restrict__ scale,
                        const float* __restrict__ shift,
                        bf16_t* __restrict__ out, unsigned total4)
{
  unsigned t = blockIdx.x * blockDim.x + threadIdx.x;
  if (t >= total4) return;
  unsigned i = t * 4u;
  int c = (int)((i / (unsigned)HW) % (unsigned)C);
  f32x4 v = *(const f32x4*)(y + i);
  float s = scale[c], b = shift[c];
  bf4 o;
#pragma unroll
  for (int j = 0; j < 4; ++j) o[j] = (bf16_t)fmaxf(fmaf(v[j], s, b), 0.0f);
  *(bf4*)(out + i) = o;
}

// im2col for 3x3 stride-2 pad-1 conv: x1 (32,128,56,56) bf16 -> col (32,1152,784)
__global__ __launch_bounds__(256)
void im2col_3x3_s2(const bf16_t* __restrict__ x, bf16_t* __restrict__ col)
{
  unsigned idx = blockIdx.x * 256u + threadIdx.x;
  const unsigned total = 32u * 1152u * 784u;
  if (idx >= total) return;
  unsigned p   = idx % 784u;
  unsigned k   = (idx / 784u) % 1152u;
  unsigned img = idx / (784u * 1152u);
  int ow = (int)(p % 28u), oh = (int)(p / 28u);
  int kw = (int)(k % 3u),  kh = (int)((k / 3u) % 3u), ci = (int)(k / 9u);
  int ih = oh * 2 - 1 + kh;
  int iw = ow * 2 - 1 + kw;
  bf16_t v = (bf16_t)0.0f;
  if ((unsigned)ih < 56u && (unsigned)iw < 56u)
    v = x[(((size_t)img * 128 + ci) * 56 + ih) * 56 + iw];
  col[idx] = v;
}

// Strided (s=2) gather of the fp32 input for the downsample conv, bf16 out.
__global__ __launch_bounds__(256)
void gather_s2(const float* __restrict__ x, bf16_t* __restrict__ out)
{
  unsigned idx = blockIdx.x * 256u + threadIdx.x;
  const unsigned total = 32u * 256u * 784u;
  if (idx >= total) return;
  unsigned p   = idx % 784u;
  unsigned ci  = (idx / 784u) % 256u;
  unsigned img = idx / (784u * 256u);
  int ow = (int)(p % 28u), oh = (int)(p / 28u);
  out[idx] = (bf16_t)x[(((size_t)img * 256 + ci) * 56 + oh * 2) * 56 + ow * 2];
}

// out = relu(bn3(y3) + bn4(y4)), 4 elements/thread.
template <int HW, int C>
__global__ __launch_bounds__(256)
void final_add_relu(const float* __restrict__ y3, const float* __restrict__ y4,
                    const float* __restrict__ sc3, const float* __restrict__ sh3,
                    const float* __restrict__ sc4, const float* __restrict__ sh4,
                    float* __restrict__ out, unsigned total4)
{
  unsigned t = blockIdx.x * blockDim.x + threadIdx.x;
  if (t >= total4) return;
  unsigned i = t * 4u;
  int c = (int)((i / (unsigned)HW) % (unsigned)C);
  f32x4 a = *(const f32x4*)(y3 + i);
  f32x4 b = *(const f32x4*)(y4 + i);
  float s3 = sc3[c], h3 = sh3[c], s4 = sc4[c], h4 = sh4[c];
  f32x4 o;
#pragma unroll
  for (int j = 0; j < 4; ++j)
    o[j] = fmaxf(fmaf(a[j], s3, h3) + fmaf(b[j], s4, h4), 0.0f);
  *(f32x4*)(out + i) = o;
}

// ---------------------------------------------------------------------------
// Launch
// ---------------------------------------------------------------------------
extern "C" void kernel_launch(void* const* d_in, const int* in_sizes, int n_in,
                              void* d_out, int out_size, void* d_ws, size_t ws_size,
                              hipStream_t stream)
{
  (void)in_sizes; (void)n_in; (void)out_size; (void)ws_size;

  const float* input = (const float*)d_in[0];
  const float* c1w   = (const float*)d_in[1];
  const float* c2w   = (const float*)d_in[2];
  const float* c3w   = (const float*)d_in[3];
  const float* c4w   = (const float*)d_in[4];
  const float* b1w   = (const float*)d_in[5];
  const float* b1b   = (const float*)d_in[6];
  const float* b2w   = (const float*)d_in[7];
  const float* b2b   = (const float*)d_in[8];
  const float* b3w   = (const float*)d_in[9];
  const float* b3b   = (const float*)d_in[10];
  const float* b4w   = (const float*)d_in[11];
  const float* b4b   = (const float*)d_in[12];

  const size_t N_Y1  = (size_t)32 * 128 * 3136;   // 12,845,056
  const size_t N_Y2  = (size_t)32 * 128 * 784;    //  3,211,264
  const size_t N_Y34 = (size_t)32 * 512 * 784;    // 12,845,056
  const size_t N_B2  = (size_t)32 * 1152 * 784;   // 28,901,376
  const size_t N_B4  = (size_t)32 * 256 * 784;    //  6,422,528

  char*  ws  = (char*)d_ws;
  size_t off = 0;
  auto alloc = [&](size_t bytes) -> void* {
    void* p = ws + off;
    off += (bytes + 255) & ~(size_t)255;
    return p;
  };
  float*  y1 = (float*)  alloc(N_Y1  * 4);
  bf16_t* x1 = (bf16_t*) alloc(N_Y1  * 2);
  bf16_t* B2 = (bf16_t*) alloc(N_B2  * 2);
  float*  y2 = (float*)  alloc(N_Y2  * 4);
  bf16_t* x2 = (bf16_t*) alloc(N_Y2  * 2);
  float*  y3 = (float*)  alloc(N_Y34 * 4);
  float*  y4 = (float*)  alloc(N_Y34 * 4);
  bf16_t* B4 = (bf16_t*) alloc(N_B4  * 2);
  float*  st = (float*)  alloc(8192  * 4);

  float *s1S = st,        *s1Q = st + 128,  *sc1 = st + 256,  *sh1 = st + 384;
  float *s2S = st + 512,  *s2Q = st + 640,  *sc2 = st + 768,  *sh2 = st + 896;
  float *s3S = st + 1024, *s3Q = st + 1536, *sc3 = st + 2048, *sh3 = st + 2560;
  float *s4S = st + 3072, *s4Q = st + 3584, *sc4 = st + 4096, *sh4 = st + 4608;

  // 0) zero the statistics accumulators and the K-split destination (y2)
  zero_kernel<<<20, 256, 0, stream>>>(st, 5120);
  zero4_kernel<<<(unsigned)(N_Y2 / 4 / 256), 256, 0, stream>>>(y2, (unsigned)(N_Y2 / 4));

  // 1) conv1 1x1 (128x256 GEMM over 3136 cols/image) + stats
  wmma_gemm_stats<float, 1><<<dim3(25, 1, 32), 256, 0, stream>>>(
      c1w, input, y1, s1S, s1Q, 256, 3136);
  finalize_bn<<<1, 128, 0, stream>>>(s1S, s1Q, b1w, b1b, sc1, sh1, 128, 1.0f / 100352.0f);
  bn_relu_store_bf16<3136, 128><<<(unsigned)(N_Y1 / 4 / 256), 256, 0, stream>>>(
      y1, sc1, sh1, x1, (unsigned)(N_Y1 / 4));

  // 2) conv2 3x3 s2: im2col, then 128x1152 GEMM split 2-ways along K
  //    (7x2x32 = 448 blocks; partials combined via f32 atomics into zeroed y2)
  im2col_3x3_s2<<<(unsigned)((N_B2 + 255) / 256), 256, 0, stream>>>(x1, B2);
  wmma_gemm_stats<bf16_t, 2><<<dim3(7, 2, 32), 256, 0, stream>>>(
      c2w, B2, y2, s2S, s2Q, 1152, 784);
  stats_from_y<<<128, 256, 0, stream>>>(y2, s2S, s2Q, 128, 784, 32);
  finalize_bn<<<1, 128, 0, stream>>>(s2S, s2Q, b2w, b2b, sc2, sh2, 128, 1.0f / 25088.0f);
  bn_relu_store_bf16<784, 128><<<(unsigned)(N_Y2 / 4 / 256), 256, 0, stream>>>(
      y2, sc2, sh2, x2, (unsigned)(N_Y2 / 4));

  // 3) conv3 1x1 (512x128)
  wmma_gemm_stats<bf16_t, 1><<<dim3(7, 4, 32), 256, 0, stream>>>(
      c3w, x2, y3, s3S, s3Q, 128, 784);

  // 4) conv4 1x1 s2 downsample (512x256) on strided input gather
  gather_s2<<<(unsigned)((N_B4 + 255) / 256), 256, 0, stream>>>(input, B4);
  wmma_gemm_stats<bf16_t, 1><<<dim3(7, 4, 32), 256, 0, stream>>>(
      c4w, B4, y4, s4S, s4Q, 256, 784);

  finalize_bn<<<4, 128, 0, stream>>>(s3S, s3Q, b3w, b3b, sc3, sh3, 512, 1.0f / 25088.0f);
  finalize_bn<<<4, 128, 0, stream>>>(s4S, s4Q, b4w, b4b, sc4, sh4, 512, 1.0f / 25088.0f);

  // 5) out = relu(bn3(y3) + bn4(y4))
  final_add_relu<784, 512><<<(unsigned)(N_Y34 / 4 / 256), 256, 0, stream>>>(
      y3, y4, sc3, sh3, sc4, sh4, (float*)d_out, (unsigned)(N_Y34 / 4));
}
